// MoELayer_9740985827631
// MI455X (gfx1250) — matrix operand order repaired and checked by reading, not compiled, not run on hardware
//
#include <hip/hip_runtime.h>
#include <math.h>
#include <stdint.h>

typedef __attribute__((ext_vector_type(16))) __bf16        v16bf;
typedef __attribute__((ext_vector_type(8)))  float         v8f;
typedef __attribute__((ext_vector_type(8)))  unsigned int  v8u;

#define Bdim 8
#define Tdim 2048
#define Ddim 512
#define Fdim 2048
#define Edim 16
#define NTOK (Bdim * Tdim)      // 16384 tokens
#define TM   32                 // tokens per FFN block
#define FCH  64                 // F-chunk per fused iteration
#define RTOK 8                  // tokens per router block (8 waves)

// dynamic LDS carve for the async FFN kernel
#define SM_XS   0                       // 32 KB  bf16 X tile
#define SM_WA   (SM_XS + TM * Ddim * 2)           // 64 KB  W1 chunk
#define SM_WB   (SM_WA + Ddim * FCH * 2)          // 64 KB  W2 chunk
#define SM_HS   (SM_WB + FCH * Ddim * 2)          //  4 KB  H tile
#define SM_B1   (SM_HS + TM * FCH * 2)            // 256 B
#define SM_TKS  (SM_B1 + FCH * 4)                 // 128 B
#define SM_GTS  (SM_TKS + TM * 4)                 // 128 B
#define SM_TOTAL (SM_GTS + TM * 4)                // 168448 B

// ---------- helpers ----------

__device__ __forceinline__ unsigned short f2bf(float f) {
  unsigned int u = __builtin_bit_cast(unsigned int, f);
  u += 0x7FFFu + ((u >> 16) & 1u);          // round-to-nearest-even
  return (unsigned short)(u >> 16);
}

__device__ __forceinline__ float gelu_tanh(float v) {
  const float c = 0.7978845608028654f;      // sqrt(2/pi)
  float u = c * (v + 0.044715f * v * v * v);
  return 0.5f * v * (1.0f + tanhf(u));
}

// Async DMA: global -> LDS, 16 bytes per active lane (GVS addressing).
__device__ __forceinline__ void async_ld_b128(unsigned lds_off, unsigned long long base,
                                              unsigned gl_off) {
  asm volatile("global_load_async_to_lds_b128 %0, %1, %2"
               :: "v"(lds_off), "v"(gl_off), "s"(base)
               : "memory");
}
// ASYNCcnt completes in issue order per wave: "<=16" retires everything older
// than the most recent 16 async instructions.
__device__ __forceinline__ void wait_async16() {
  asm volatile("s_wait_asynccnt 16" ::: "memory");
}
__device__ __forceinline__ void wait_async0() {
  asm volatile("s_wait_asynccnt 0" ::: "memory");
}

// A-matrix fragment (16-bit, 16xK=32), row-major LDS tile, stride in elements.
__device__ __forceinline__ v16bf lds_load_a(const unsigned short* base, int m0, int k0, int stride) {
  int lane = threadIdx.x & 31;
  int m  = m0 + (lane & 15);
  int hi = lane >> 4;
  const unsigned int* p = (const unsigned int*)base;
  v8u u;
#pragma unroll
  for (int v = 0; v < 4; ++v)
    u[v] = p[(m * stride + k0 + 2 * v + 8 * hi) >> 1];
#pragma unroll
  for (int v = 4; v < 8; ++v)
    u[v] = p[(m * stride + k0 + 16 + 2 * (v - 4) + 8 * hi) >> 1];
  return __builtin_bit_cast(v16bf, u);
}

// B-matrix fragment (16-bit, K=32 x 16), row-major [K][N] LDS tile.
__device__ __forceinline__ v16bf lds_load_b(const unsigned short* base, int k0, int n0, int stride) {
  int lane = threadIdx.x & 31;
  int n  = n0 + (lane & 15);
  int hi = lane >> 4;
  v8u u;
#pragma unroll
  for (int v = 0; v < 8; ++v) {
    int k = k0 + ((v < 4) ? (2 * v) : (16 + 2 * (v - 4))) + 8 * hi;
    u[v] = (unsigned)base[k * stride + n] | ((unsigned)base[(k + 1) * stride + n] << 16);
  }
  return __builtin_bit_cast(v16bf, u);
}

// ---------- fp32 -> packed bf16 conversion ----------

__global__ __launch_bounds__(256) void cvt_bf16_pack(const float* __restrict__ src,
                                                     unsigned int* __restrict__ dst, int n2) {
  for (int i = blockIdx.x * 256 + threadIdx.x; i < n2; i += gridDim.x * 256) {
    float a = src[2 * i], b = src[2 * i + 1];
    dst[i] = (unsigned)f2bf(a) | ((unsigned)f2bf(b) << 16);
  }
}

// ---------- router ----------

__global__ __launch_bounds__(256) void router_kernel(
    const float* __restrict__ x, const float* __restrict__ Wr,
    int* __restrict__ ecnt, int* __restrict__ etok, float* __restrict__ egate,
    float* __restrict__ partials) {
  __shared__ float wr_s[Ddim * Edim];     // 32 KB
  __shared__ float psh[RTOK][Edim];
  __shared__ float csh[RTOK][Edim];

  int tid = threadIdx.x;
  for (int i = tid; i < Ddim * Edim; i += 256) wr_s[i] = Wr[i];
  __syncthreads();

  int wave = tid >> 5, lane = tid & 31;
  int tok  = blockIdx.x * RTOK + wave;

  float acc[Edim];
#pragma unroll
  for (int e = 0; e < Edim; ++e) acc[e] = 0.0f;

  const float* xr = x + (size_t)tok * Ddim;
#pragma unroll 4
  for (int i = 0; i < Ddim / 32; ++i) {
    int d = lane + 32 * i;
    float xv = xr[d];
#pragma unroll
    for (int e = 0; e < Edim; ++e) acc[e] += xv * wr_s[d * Edim + e];
  }
#pragma unroll
  for (int e = 0; e < Edim; ++e) {
    float v = acc[e];
#pragma unroll
    for (int off = 16; off >= 1; off >>= 1) v += __shfl_xor(v, off, 32);
    acc[e] = v;
  }

  if (lane == 0) {
    float mx = acc[0];
#pragma unroll
    for (int e = 1; e < Edim; ++e) mx = fmaxf(mx, acc[e]);
    float p[Edim], s = 0.0f;
#pragma unroll
    for (int e = 0; e < Edim; ++e) { p[e] = __expf(acc[e] - mx); s += p[e]; }
    float inv = 1.0f / s;
#pragma unroll
    for (int e = 0; e < Edim; ++e) psh[wave][e] = p[e] * inv;

    int i0 = 0; float v0 = acc[0];
#pragma unroll
    for (int e = 1; e < Edim; ++e) if (acc[e] > v0) { v0 = acc[e]; i0 = e; }
    int i1 = (i0 == 0) ? 1 : 0; float v1 = acc[i1];
#pragma unroll
    for (int e = 0; e < Edim; ++e)
      if (e != i0 && acc[e] > v1) { v1 = acc[e]; i1 = e; }

    float t  = __expf(v1 - v0);
    float g0 = 1.0f / (1.0f + t);
    float g1 = t / (1.0f + t);

    int p0 = atomicAdd(&ecnt[i0], 1);
    etok[i0 * NTOK + p0] = tok;  egate[i0 * NTOK + p0] = g0;
    int p1 = atomicAdd(&ecnt[i1], 1);
    etok[i1 * NTOK + p1] = tok;  egate[i1 * NTOK + p1] = g1;

#pragma unroll
    for (int e = 0; e < Edim; ++e) csh[wave][e] = (e == i0) ? 1.0f : 0.0f;
  }
  __syncthreads();

  if (tid < Edim) {
    float sp = 0.0f, sc = 0.0f;
#pragma unroll
    for (int w = 0; w < RTOK; ++w) { sp += psh[w][tid]; sc += csh[w][tid]; }
    partials[blockIdx.x * 32 + tid]        = sp;
    partials[blockIdx.x * 32 + Edim + tid] = sc;
  }
}

// ---------- aux loss reduction ----------

__global__ void aux_kernel(const float* __restrict__ partials, int nblocks,
                           float* __restrict__ out_aux) {
  __shared__ float fP[Edim], fC[Edim];
  int e = threadIdx.x;
  if (e < Edim) {
    float sp = 0.0f, sc = 0.0f;
    for (int b = 0; b < nblocks; ++b) {
      sp += partials[b * 32 + e];
      sc += partials[b * 32 + Edim + e];
    }
    fP[e] = sp; fC[e] = sc;
  }
  __syncthreads();
  if (threadIdx.x == 0) {
    const float invN = 1.0f / (float)NTOK;
    float aux = 0.0f;
    for (int k = 0; k < Edim; ++k) aux += (fC[k] * invN) * (fP[k] * invN);
    *out_aux = (float)Edim * aux;
  }
}

// ---------- fused FFN: double-buffered async DMA pipeline ----------

__global__ __launch_bounds__(256) void ffn_kernel_async(
    const unsigned short* __restrict__ xbf,
    const unsigned short* __restrict__ w1bf, const float* __restrict__ b1,
    const unsigned short* __restrict__ w2bf, const float* __restrict__ b2,
    const int* __restrict__ ecnt, const int* __restrict__ etok,
    const float* __restrict__ egate, float* __restrict__ out) {
  int e   = blockIdx.y;
  int cnt = ecnt[e];
  int tb  = blockIdx.x * TM;
  if (tb >= cnt) return;

  extern __shared__ __align__(16) unsigned char smem[];
  unsigned short* Xs  = (unsigned short*)(smem + SM_XS);
  unsigned short* WA  = (unsigned short*)(smem + SM_WA);   // W1 chunks
  unsigned short* WB  = (unsigned short*)(smem + SM_WB);   // W2 chunks
  unsigned short* Hs  = (unsigned short*)(smem + SM_HS);
  float*          b1s = (float*)(smem + SM_B1);
  int*            tks = (int*)(smem + SM_TKS);
  float*          gts = (float*)(smem + SM_GTS);

  int tid = threadIdx.x;
  if (tid < TM) {
    int idx = tb + tid;
    tks[tid] = (idx < cnt) ? etok[e * NTOK + idx] : 0;
    gts[tid] = (idx < cnt) ? egate[e * NTOK + idx] : 0.0f;
  }
  __syncthreads();

  unsigned ldsX = (unsigned)(uintptr_t)Xs;
  unsigned ldsA = (unsigned)(uintptr_t)WA;
  unsigned ldsB = (unsigned)(uintptr_t)WB;
  unsigned long long xb  = (unsigned long long)xbf;
  unsigned long long w1b = (unsigned long long)(w1bf + (size_t)e * Ddim * Fdim);
  unsigned long long w2b = (unsigned long long)(w2bf + (size_t)e * Fdim * Ddim);

  // ---- prologue: X gather (8 instrs/wave) + W1[0] (16 instrs/wave) ----
#pragma unroll
  for (int it = 0; it < 8; ++it) {
    int u = tid + it * 256;           // 16-byte unit; 64 units per 1 KB row
    int r = u >> 6, c = u & 63;
    async_ld_b128(ldsX + u * 16, xb, (unsigned)(tks[r] * (Ddim * 2) + c * 16));
  }
#pragma unroll
  for (int it = 0; it < 16; ++it) {
    int u = tid + it * 256;           // [d][j]: 8 units per 64-col row
    int d = u >> 3, j16 = u & 7;
    async_ld_b128(ldsA + u * 16, w1b, (unsigned)(d * (Fdim * 2) + j16 * 16));
  }

  int wave = tid >> 5;
  int lane = tid & 31;
  int wr = wave & 1;        // row tile (2 x 16 rows)
  int wc = wave >> 1;       // col group (4 x 128 cols)

  v8f oacc[8];
#pragma unroll
  for (int ct = 0; ct < 8; ++ct)
#pragma unroll
    for (int v = 0; v < 8; ++v) oacc[ct][v] = 0.0f;

  for (int fc = 0; fc < Fdim; fc += FCH) {
    // (1) kick W2[fc] -> WB; overlaps with GEMM1 below
#pragma unroll
    for (int it = 0; it < 16; ++it) {
      int u = tid + it * 256;         // [f][d]: 64 units per 512-col row
      int f = u >> 6, t16 = u & 63;
      async_ld_b128(ldsB + u * 16, w2b, (unsigned)((fc + f) * (Ddim * 2) + t16 * 16));
    }
    if (tid < FCH) b1s[tid] = b1[e * Fdim + fc + tid];

    // (2) oldest batches (X + W1[fc]) retired; W2[fc] may still fly
    wait_async16();
    __syncthreads();

    // (3) GEMM1: each wave one 16x16 H tile; contraction over D=512
    {
      int hr = wave & 1, hc = wave >> 1;
      v8f hacc;
#pragma unroll
      for (int v = 0; v < 8; ++v) hacc[v] = 0.0f;
#pragma unroll 4
      for (int kk = 0; kk < Ddim; kk += 32) {
        v16bf a  = lds_load_a(Xs, hr * 16, kk, Ddim);
        v16bf bB = lds_load_b(WA, kk, hc * 16, FCH);
        hacc = __builtin_amdgcn_wmma_f32_16x16x32_bf16(
            false, a, false, bB, (short)0, hacc, false, false);
      }
      int n  = lane & 15;
      int hi = lane >> 4;
      float bv = b1s[hc * 16 + n];
#pragma unroll
      for (int v = 0; v < 8; ++v) {
        float h = gelu_tanh(hacc[v] + bv);
        int m = hr * 16 + v + 8 * hi;
        Hs[m * FCH + hc * 16 + n] = f2bf(h);
      }
    }
    __syncthreads();   // H ready; all waves done reading WA

    // (4) kick W1[fc+64] -> WA; overlaps with GEMM2 below
    if (fc + FCH < Fdim) {
#pragma unroll
      for (int it = 0; it < 16; ++it) {
        int u = tid + it * 256;
        int d = u >> 3, j16 = u & 7;
        async_ld_b128(ldsA + u * 16, w1b,
                      (unsigned)(d * (Fdim * 2) + (fc + FCH) * 2 + j16 * 16));
      }
      wait_async16();   // W2[fc] (older) retired; W1[fc+64] may still fly
    } else {
      wait_async0();    // nothing younger: drain so W2[fc] is resident
    }
    __syncthreads();

    // (5) GEMM2: wave accumulates its 16x128 output strip
#pragma unroll
    for (int kk = 0; kk < FCH; kk += 32) {
      v16bf a = lds_load_a(Hs, wr * 16, kk, FCH);
#pragma unroll
      for (int ct = 0; ct < 8; ++ct) {
        v16bf bB = lds_load_b(WB, kk, wc * 128 + ct * 16, Ddim);
        oacc[ct] = __builtin_amdgcn_wmma_f32_16x16x32_bf16(
            false, a, false, bB, (short)0, oacc[ct], false, false);
      }
    }
    __syncthreads();   // all waves done reading WB before next W2 DMA
  }

  int n  = lane & 15;
  int hi = lane >> 4;
#pragma unroll
  for (int ct = 0; ct < 8; ++ct) {
    int dcol = wc * 128 + ct * 16 + n;
    float b2v = b2[e * Ddim + dcol];
#pragma unroll
    for (int v = 0; v < 8; ++v) {
      int m = wr * 16 + v + 8 * hi;
      if (tb + m < cnt) {
        float val = gts[m] * (oacc[ct][v] + b2v);
        atomicAdd(&out[(size_t)tks[m] * Ddim + dcol], val);
      }
    }
  }
}

// ---------- fallback FFN (fp32 global, convert while staging) ----------

__global__ __launch_bounds__(256) void ffn_kernel_f32(
    const float* __restrict__ x,
    const float* __restrict__ W1, const float* __restrict__ b1,
    const float* __restrict__ W2, const float* __restrict__ b2,
    const int* __restrict__ ecnt, const int* __restrict__ etok,
    const float* __restrict__ egate, float* __restrict__ out) {
  int e   = blockIdx.y;
  int cnt = ecnt[e];
  int tb  = blockIdx.x * TM;
  if (tb >= cnt) return;

  __shared__ __align__(16) unsigned short Xs[TM * Ddim];
  __shared__ __align__(16) unsigned short Ws[Ddim * FCH];
  __shared__ __align__(16) unsigned short Hs[TM * FCH];
  __shared__ float b1s[FCH];
  __shared__ int   tks[TM];
  __shared__ float gts[TM];

  int tid = threadIdx.x;
  if (tid < TM) {
    int idx = tb + tid;
    tks[tid] = (idx < cnt) ? etok[e * NTOK + idx] : 0;
    gts[tid] = (idx < cnt) ? egate[e * NTOK + idx] : 0.0f;
  }
  __syncthreads();

  for (int i = tid; i < TM * Ddim; i += 256) {
    int r = i >> 9, c = i & (Ddim - 1);
    Xs[i] = f2bf(x[(size_t)tks[r] * Ddim + c]);
  }

  const float* W1e = W1 + (size_t)e * Ddim * Fdim;
  const float* W2e = W2 + (size_t)e * Fdim * Ddim;

  int wave = tid >> 5;
  int lane = tid & 31;
  int wr = wave & 1;
  int wc = wave >> 1;

  v8f oacc[8];
#pragma unroll
  for (int ct = 0; ct < 8; ++ct)
#pragma unroll
    for (int v = 0; v < 8; ++v) oacc[ct][v] = 0.0f;

  for (int fc = 0; fc < Fdim; fc += FCH) {
    __syncthreads();
    for (int i = tid; i < Ddim * FCH; i += 256) {
      int d = i >> 6, j = i & (FCH - 1);
      Ws[i] = f2bf(W1e[(size_t)d * Fdim + fc + j]);
    }
    if (tid < FCH) b1s[tid] = b1[e * Fdim + fc + tid];
    if (fc + FCH < Fdim)
      __builtin_prefetch(&W1e[(size_t)(tid * 2) * Fdim + fc + FCH], 0, 0);
    __syncthreads();

    {
      int hr = wave & 1, hc = wave >> 1;
      v8f hacc;
#pragma unroll
      for (int v = 0; v < 8; ++v) hacc[v] = 0.0f;
#pragma unroll 4
      for (int kk = 0; kk < Ddim; kk += 32) {
        v16bf a  = lds_load_a(Xs, hr * 16, kk, Ddim);
        v16bf bB = lds_load_b(Ws, kk, hc * 16, FCH);
        hacc = __builtin_amdgcn_wmma_f32_16x16x32_bf16(
            false, a, false, bB, (short)0, hacc, false, false);
      }
      int n  = lane & 15;
      int hi = lane >> 4;
      float bv = b1s[hc * 16 + n];
#pragma unroll
      for (int v = 0; v < 8; ++v) {
        float h = gelu_tanh(hacc[v] + bv);
        int m = hr * 16 + v + 8 * hi;
        Hs[m * FCH + hc * 16 + n] = f2bf(h);
      }
    }
    __syncthreads();

    for (int i = tid; i < FCH * Ddim; i += 256) {
      int f = i >> 9, dcol = i & (Ddim - 1);
      Ws[i] = f2bf(W2e[(size_t)(fc + f) * Ddim + dcol]);
    }
    __syncthreads();

#pragma unroll
    for (int kk = 0; kk < FCH; kk += 32) {
      v16bf a = lds_load_a(Hs, wr * 16, kk, FCH);
#pragma unroll
      for (int ct = 0; ct < 8; ++ct) {
        v16bf bB = lds_load_b(Ws, kk, wc * 128 + ct * 16, Ddim);
        oacc[ct] = __builtin_amdgcn_wmma_f32_16x16x32_bf16(
            false, a, false, bB, (short)0, oacc[ct], false, false);
      }
    }
  }

  int n  = lane & 15;
  int hi = lane >> 4;
#pragma unroll
  for (int ct = 0; ct < 8; ++ct) {
    int dcol = wc * 128 + ct * 16 + n;
    float b2v = b2[e * Ddim + dcol];
#pragma unroll
    for (int v = 0; v < 8; ++v) {
      int m = wr * 16 + v + 8 * hi;
      if (tb + m < cnt) {
        float val = gts[m] * (oacc[ct][v] + b2v);
        atomicAdd(&out[(size_t)tks[m] * Ddim + dcol], val);
      }
    }
  }
}

// ---------- launch ----------

extern "C" void kernel_launch(void* const* d_in, const int* in_sizes, int n_in,
                              void* d_out, int out_size, void* d_ws, size_t ws_size,
                              hipStream_t stream) {
  const float* x  = (const float*)d_in[0];
  const float* Wr = (const float*)d_in[1];
  const float* W1 = (const float*)d_in[2];
  const float* b1 = (const float*)d_in[3];
  const float* W2 = (const float*)d_in[4];
  const float* b2 = (const float*)d_in[5];
  float* out = (float*)d_out;

  const int nRouterBlocks = NTOK / RTOK;       // 2048
  char* ws = (char*)d_ws;

  size_t offEcnt = 0;                                    // 16 ints
  size_t offPart = 256;                                  // 2048*32 f32
  size_t offEtok = offPart + (size_t)nRouterBlocks * 32 * 4;
  size_t offEgat = offEtok + (size_t)Edim * NTOK * 4;
  size_t offXbf  = (offEgat + (size_t)Edim * NTOK * 4 + 255) & ~(size_t)255;
  size_t offW1bf = offXbf  + (size_t)NTOK * Ddim * 2;
  size_t offW2bf = offW1bf + (size_t)Edim * Ddim * Fdim * 2;
  size_t offEnd  = offW2bf + (size_t)Edim * Fdim * Ddim * 2;

  int*   ecnt     = (int*)(ws + offEcnt);
  float* partials = (float*)(ws + offPart);
  int*   etok     = (int*)(ws + offEtok);
  float* egate    = (float*)(ws + offEgat);

  hipMemsetAsync(ecnt, 0, 64, stream);
  hipMemsetAsync(d_out, 0, (size_t)out_size * sizeof(float), stream);

  router_kernel<<<nRouterBlocks, 256, 0, stream>>>(x, Wr, ecnt, etok, egate, partials);
  aux_kernel<<<1, 32, 0, stream>>>(partials, nRouterBlocks, out + (size_t)NTOK * Ddim);

  dim3 grid(NTOK / TM, Edim);
  if (ws_size >= offEnd) {
    unsigned short* xbf  = (unsigned short*)(ws + offXbf);
    unsigned short* w1bf = (unsigned short*)(ws + offW1bf);
    unsigned short* w2bf = (unsigned short*)(ws + offW2bf);
    int nx = NTOK * Ddim / 2, nw = Edim * Ddim * Fdim / 2;
    cvt_bf16_pack<<<4096, 256, 0, stream>>>(x,  (unsigned int*)xbf,  nx);
    cvt_bf16_pack<<<8192, 256, 0, stream>>>(W1, (unsigned int*)w1bf, nw);
    cvt_bf16_pack<<<8192, 256, 0, stream>>>(W2, (unsigned int*)w2bf, nw);
    ffn_kernel_async<<<grid, 256, SM_TOTAL, stream>>>(xbf, w1bf, b1, w2bf, b2,
                                                      ecnt, etok, egate, out);
  } else {
    ffn_kernel_f32<<<grid, 256, 0, stream>>>(x, W1, b1, W2, b2,
                                             ecnt, etok, egate, out);
  }
}
